// FTTransformerPNAFusedLayer_83588653515397
// MI455X (gfx1250) — matrix operand order repaired and checked by reading, not compiled, not run on hardware
//
#include <hip/hip_runtime.h>
#include <math.h>
#include <stdint.h>

// ---------------- problem constants ----------------
#define B_     4096
#define S_     32
#define C_     128
#define H_     8
#define NH_    128
#define NN_    20000
#define EE_    320000
#define FUSED_ 384
#define MTAB_  (B_ * S_)      // 131072

typedef __attribute__((ext_vector_type(16))) _Float16 v16h;
typedef __attribute__((ext_vector_type(8)))  float    v8f;
typedef __attribute__((ext_vector_type(4)))  unsigned int u32x4;
typedef __attribute__((ext_vector_type(8)))  int          i32x8;
typedef __attribute__((ext_vector_type(4)))  int          i32x4;

#if defined(__has_builtin)
#  if __has_builtin(__builtin_amdgcn_tensor_load_to_lds) && \
      __has_builtin(__builtin_amdgcn_s_wait_tensorcnt)
#    define HAS_TDM 1
#  endif
#endif
#ifndef HAS_TDM
#  define HAS_TDM 0
#endif

#define BPAD 36   // LDS row stride (floats) for a 32-float row + 4 pad dwords

// ---------------- WMMA fragment helpers (ISA 7.12.2 layouts, wave32) -------
// A: 16x32 f16 from row-major f32 source (lda = K).
__device__ inline v16h load_a_frag(const float* __restrict__ A, int lda) {
  const int lane = threadIdx.x & 31;
  const int m = lane & 15;
  const int kh = (lane >> 4) << 3;       // 0 or 8
  v16h a;
#pragma unroll
  for (int i = 0; i < 16; ++i) {
    int k = kh + (i & 7) + ((i >> 3) << 4);
    a[i] = (_Float16)A[m * lda + k];
  }
  return a;
}

// B: 32x16 f16 built from row-major W[N][K] in global memory (X @ W^T).
__device__ inline v16h load_b_frag_wt(const float* __restrict__ W, int ldw) {
  const int lane = threadIdx.x & 31;
  const int n = lane & 15;
  const int kb = (lane >> 4) << 3;       // 0 or 8
  v16h b;
#pragma unroll
  for (int i = 0; i < 16; ++i)
    b[i] = (_Float16)W[n * ldw + kb + (i & 7) + ((i >> 3) << 4)];
  return b;
}

// B: 32x16 f16 from an LDS slab with padded row stride BPAD.
__device__ inline v16h load_b_frag_lds(const float* slab) {
  const int lane = threadIdx.x & 31;
  const int n = lane & 15;
  const int kb = (lane >> 4) << 3;
  v16h b;
#pragma unroll
  for (int i = 0; i < 16; ++i)
    b[i] = (_Float16)slab[n * BPAD + kb + (i & 7) + ((i >> 3) << 4)];
  return b;
}

#if HAS_TDM
// Issue a TDM load of a 64(N) x 32(K) f32 tile of row-major W (row stride K
// elements) into LDS at lds_off, padding 4 dwords after every 32-dword row
// (so the LDS row stride becomes BPAD floats).  D# per ISA 08_async_tensor.
__device__ inline void tdm_load_b_slab(const float* Wtile, int K, unsigned lds_off) {
  uint64_t ga = (uint64_t)(uintptr_t)Wtile;
  u32x4 g0;
  g0[0] = 1u;                                            // count=1, user mode
  g0[1] = lds_off;                                       // lds_addr
  g0[2] = (unsigned)ga;                                  // global_addr lo
  g0[3] = ((unsigned)(ga >> 32) & 0x01FFFFFFu) | (2u << 30);  // addr hi | type=2
  i32x8 g1;
  g1[0] = (int)((2u << 16) | (1u << 20) | (4u << 22) | (3u << 25));
         // data_size=4B | pad_enable | pad_interval=32dw | pad_amount=4dw
  g1[1] = (int)(((unsigned)K & 0xFFFFu) << 16);          // tensor_dim0 lo16
  g1[2] = (int)((((unsigned)K >> 16) & 0xFFFFu) | (64u << 16)); // dim0 hi | tensor_dim1=64
  g1[3] = (int)(32u << 16);                              // tile_dim0 = 32
  g1[4] = 64;                                            // tile_dim1 = 64
  g1[5] = (int)(unsigned)K;                              // tensor_dim0_stride lo32
  g1[6] = 0;
  g1[7] = 0;
  i32x4 z4 = {0, 0, 0, 0};
#if __clang_major__ >= 23
  i32x8 z8 = {0, 0, 0, 0, 0, 0, 0, 0};
  __builtin_amdgcn_tensor_load_to_lds(g0, g1, z4, z4, z8, 0);
#else
  __builtin_amdgcn_tensor_load_to_lds(g0, g1, z4, z4, 0);
#endif
}
#endif

// ---------------- float atomic max/min via CAS -----------------------------
__device__ inline void atomicMaxF(float* addr, float val) {
  unsigned int* ua = (unsigned int*)addr;
  unsigned int old = *ua;
  while (__uint_as_float(old) < val) {
    unsigned int assumed = old;
    old = atomicCAS(ua, assumed, __float_as_uint(val));
    if (old == assumed) break;
  }
}
__device__ inline void atomicMinF(float* addr, float val) {
  unsigned int* ua = (unsigned int*)addr;
  unsigned int old = *ua;
  while (__uint_as_float(old) > val) {
    unsigned int assumed = old;
    old = atomicCAS(ua, assumed, __float_as_uint(val));
    if (old == assumed) break;
  }
}

// ---------------- generic WMMA GEMM:  Y = epi(A @ W^T + bias [+ R]) --------
// A: [M x K] row-major, W: [N x K] row-major, Y/R: [M x N] row-major.
// Block = 4 waves computing a 64x64 tile (each wave 16 rows x 64 cols,
// 4 accumulators, A fragment reused 4x).  B slab double-buffered in LDS via
// the Tensor Data Mover (fallback: cooperative ds staging).
// Requires: N % 64 == 0, K % 32 == 0.  M handled with clamping.
enum { EPI_BIAS = 0, EPI_RELU = 1, EPI_LEAKY = 2, EPI_RES = 3 };

template <int EPI>
__global__ __launch_bounds__(128) void gemm_wmma_kernel(
    const float* __restrict__ A, const float* __restrict__ W,
    const float* __restrict__ bias, const float* __restrict__ R,
    float* __restrict__ Y, int M, int N, int K) {
  __shared__ float sb[2][64 * BPAD];
  const int nblk = N >> 6;
  const int mt64 = blockIdx.x / nblk;
  const int nt64 = blockIdx.x % nblk;
  const int wave = threadIdx.x >> 5;
  const int lane = threadIdx.x & 31;
  const int n0 = nt64 << 6;
  int m0 = (mt64 << 6) + (wave << 4);
  const bool valid = (m0 + 16 <= M);
  if (!valid) m0 = M - 16;                 // clamp to stay in bounds
  const float* Ab = A + (size_t)m0 * K;
  const float* Wn = W + (size_t)n0 * K;

  v8f z = {};
  v8f acc[4] = {z, z, z, z};

#if HAS_TDM
  if (wave == 0) tdm_load_b_slab(Wn, K, (unsigned)(uintptr_t)&sb[0][0]);
#endif
  for (int k0 = 0, it = 0; k0 < K; k0 += 32, ++it) {
    const int cur = it & 1;
#if HAS_TDM
    __syncthreads();                       // readers of sb[cur^1] finished
    if (wave == 0) {
      if (k0 + 32 < K) {
        tdm_load_b_slab(Wn + (k0 + 32), K, (unsigned)(uintptr_t)&sb[cur ^ 1][0]);
        __builtin_amdgcn_s_wait_tensorcnt(1);   // slab[cur] complete
      } else {
        __builtin_amdgcn_s_wait_tensorcnt(0);
      }
    }
    __syncthreads();                       // publish slab[cur]
#else
    __syncthreads();
    for (int idx = threadIdx.x; idx < 64 * 32; idx += 128) {
      int r = idx >> 5, ci = idx & 31;
      sb[cur][r * BPAD + ci] = Wn[(size_t)r * K + k0 + ci];
    }
    __syncthreads();
#endif
    const float* slab = &sb[cur][0];
    v16h a = load_a_frag(Ab + k0, K);
#pragma unroll
    for (int t = 0; t < 4; ++t) {
      v16h b = load_b_frag_lds(slab + t * 16 * BPAD);
      acc[t] = __builtin_amdgcn_wmma_f32_16x16x32_f16(false, a, false, b,
                                                      (short)0, acc[t], false, false);
    }
  }

  if (valid) {
    const int nc = lane & 15;
    const int mrow = m0 + ((lane >> 4) << 3);
#pragma unroll
    for (int t = 0; t < 4; ++t) {
      const int n = n0 + t * 16 + nc;
      const float bv = bias ? bias[n] : 0.f;
#pragma unroll
      for (int r = 0; r < 8; ++r) {
        float v = acc[t][r] + bv;
        if (EPI == EPI_RELU)  v = fmaxf(v, 0.f);
        if (EPI == EPI_LEAKY) v = v > 0.f ? v : 0.01f * v;
        if (EPI == EPI_RES)   v += R[(size_t)(mrow + r) * N + n];
        Y[(size_t)(mrow + r) * N + n] = v;
      }
    }
  }
}

// ---------------- message GEMM fused with segment reductions ---------------
// msg = concat[x_gnn[dst], x_gnn[src], e] @ Wpre^T + bpre   (E x 128, K=384)
// One wave per 16x64 strip (4 accumulators, gathered A fragment reused 4x).
// Epilogue scatters sum / sum^2 / max / min into per-node stats via atomics.
__global__ __launch_bounds__(128) void gemm_msg_kernel(
    const float* __restrict__ xg, const int* __restrict__ srcI,
    const int* __restrict__ dstI, const float* __restrict__ e,
    const float* __restrict__ Wpre, const float* __restrict__ bpre,
    float* __restrict__ sumb, float* __restrict__ sqb,
    float* __restrict__ mxb, float* __restrict__ mnb) {
  const int nstrips = NH_ >> 6;            // 2
  int wv = blockIdx.x * (blockDim.x >> 5) + (threadIdx.x >> 5);
  int mt = wv / nstrips, nt64 = wv % nstrips;
  if (mt * 16 >= EE_) return;
  const int lane = threadIdx.x & 31;
  const int am = mt * 16 + (lane & 15);
  const int kh = (lane >> 4) << 3;
  const float* prow_d = xg + (size_t)dstI[am] * NH_;
  const float* prow_s = xg + (size_t)srcI[am] * NH_;
  const float* prow_e = e + (size_t)am * NH_;

  v8f z = {};
  v8f acc[4] = {z, z, z, z};
  for (int k0 = 0; k0 < 384; k0 += 32) {
    v16h a;
#pragma unroll
    for (int i = 0; i < 16; ++i) {
      int k = k0 + kh + (i & 7) + ((i >> 3) << 4);
      float f = (k < 128) ? prow_d[k]
              : (k < 256) ? prow_s[k - 128]
                          : prow_e[k - 256];
      a[i] = (_Float16)f;
    }
#pragma unroll
    for (int t = 0; t < 4; ++t) {
      const float* Wb = Wpre + (size_t)(nt64 * 64 + t * 16) * 384;
      v16h b = load_b_frag_wt(Wb + k0, 384);
      acc[t] = __builtin_amdgcn_wmma_f32_16x16x32_f16(false, a, false, b,
                                                      (short)0, acc[t], false, false);
    }
  }
  const int nc = lane & 15;
  const int mb = mt * 16 + ((lane >> 4) << 3);
#pragma unroll
  for (int t = 0; t < 4; ++t) {
    const int n = nt64 * 64 + t * 16 + nc;
    const float bv = bpre[n];
#pragma unroll
    for (int r = 0; r < 8; ++r) {
      int node = dstI[mb + r];
      float v = acc[t][r] + bv;
      size_t idx = (size_t)node * NH_ + n;
      atomicAdd(&sumb[idx], v);
      atomicAdd(&sqb[idx], v * v);
      atomicMaxF(&mxb[idx], v);
      atomicMinF(&mnb[idx], v);
    }
  }
}

// ---------------- attention (B*H blocks of one wave; S=32, hd=16) ----------
__global__ void attn_kernel(const float* __restrict__ qkv, float* __restrict__ o) {
  int b = blockIdx.x >> 3, h = blockIdx.x & 7;
  __shared__ float ks[32][16], vs[32][16];
  int lane = threadIdx.x;                 // query row
  const float* base = qkv + (size_t)b * S_ * (3 * C_);
  float q[16];
#pragma unroll
  for (int d = 0; d < 16; ++d) {
    q[d]        = base[lane * 384 + h * 16 + d];
    ks[lane][d] = base[lane * 384 + 128 + h * 16 + d];
    vs[lane][d] = base[lane * 384 + 256 + h * 16 + d];
  }
  __syncthreads();
  float s[32], mx = -3.402823466e38f;
  for (int j = 0; j < 32; ++j) {
    float acc = 0.f;
#pragma unroll
    for (int d = 0; d < 16; ++d) acc += q[d] * ks[j][d];
    acc *= 0.25f;                         // 1/sqrt(16)
    s[j] = acc; mx = fmaxf(mx, acc);
  }
  float sum = 0.f;
  for (int j = 0; j < 32; ++j) { s[j] = __expf(s[j] - mx); sum += s[j]; }
  float inv = 1.f / sum;
  float acc[16];
#pragma unroll
  for (int d = 0; d < 16; ++d) acc[d] = 0.f;
  for (int j = 0; j < 32; ++j) {
    float p = s[j] * inv;
#pragma unroll
    for (int d = 0; d < 16; ++d) acc[d] += p * vs[j][d];
  }
  float* ob = o + ((size_t)b * S_ + lane) * C_ + h * 16;
#pragma unroll
  for (int d = 0; d < 16; ++d) ob[d] = acc[d];
}

// ---------------- LayerNorm: one wave per row ------------------------------
__global__ void ln_kernel(const float* __restrict__ X, const float* __restrict__ w,
                          const float* __restrict__ b, float* __restrict__ Y,
                          int rows, int width) {
  int row = blockIdx.x * (blockDim.x >> 5) + (threadIdx.x >> 5);
  if (row >= rows) return;
  int lane = threadIdx.x & 31;
  const float* x = X + (size_t)row * width;
  float s = 0.f;
  for (int i = lane; i < width; i += 32) s += x[i];
  for (int off = 16; off > 0; off >>= 1) s += __shfl_xor(s, off, 32);
  float mean = s / width;
  float v = 0.f;
  for (int i = lane; i < width; i += 32) { float d = x[i] - mean; v += d * d; }
  for (int off = 16; off > 0; off >>= 1) v += __shfl_xor(v, off, 32);
  float rstd = rsqrtf(v / width + 1e-5f);
  float* y = Y + (size_t)row * width;
  for (int i = lane; i < width; i += 32)
    y[i] = (x[i] - mean) * rstd * w[i] + b[i];
}

// ---------------- small utility kernels ------------------------------------
__global__ void copy_kernel(const float* __restrict__ s, float* __restrict__ d, size_t n) {
  size_t i = (size_t)blockIdx.x * blockDim.x + threadIdx.x;
  size_t str = (size_t)gridDim.x * blockDim.x;
  for (; i < n; i += str) d[i] = s[i];
}

__global__ void cls_extract_kernel(const float* __restrict__ xtab, float* __restrict__ cls) {
  cls[(size_t)blockIdx.x * C_ + threadIdx.x] =
      xtab[(size_t)blockIdx.x * S_ * C_ + threadIdx.x];
}

__global__ void init_stats_kernel(float* sum, float* sq, float* mx, float* mn,
                                  float* cnt, float* avg, float* bns, float* bnq) {
  int i = blockIdx.x * blockDim.x + threadIdx.x;
  if (i < NN_ * NH_) {
    sum[i] = 0.f; sq[i] = 0.f;
    mx[i] = -3.402823466e38f; mn[i] = 3.402823466e38f;
  }
  if (i < NN_) cnt[i] = 0.f;
  if (i == 0) avg[0] = 0.f;
  if (i < NH_) { bns[i] = 0.f; bnq[i] = 0.f; }
}

__global__ void count_kernel(const int* __restrict__ dstI, float* __restrict__ cnt) {
  int i = blockIdx.x * blockDim.x + threadIdx.x;
  if (i < EE_) atomicAdd(&cnt[dstI[i]], 1.f);
}

__global__ void avg_log_kernel(const float* __restrict__ cnt, float* __restrict__ avg) {
  int i = blockIdx.x * blockDim.x + threadIdx.x;
  if (i < NN_) atomicAdd(avg, logf(cnt[i] + 1.f));
}

// Build post-aggregation feature rows: [x_gnn | agg | agg*logd/avg | agg*avg/logd]
__global__ void feat_kernel(const float* __restrict__ xg, const float* __restrict__ sumb,
                            const float* __restrict__ sqb, const float* __restrict__ mxb,
                            const float* __restrict__ mnb, const float* __restrict__ cntb,
                            const float* __restrict__ avg_acc, float* __restrict__ feat) {
  int n = blockIdx.x, t = threadIdx.x;              // 128 threads
  float cnt = cntb[n];
  float denom = fmaxf(cnt, 1.f);
  bool has = cnt > 0.f;
  float avg = avg_acc[0] / (float)NN_;
  float logd = logf(denom + 1.f);
  float r1 = logd / avg, r2 = avg / logd;
  float* fr = feat + (size_t)n * (13 * NH_);
  fr[t] = xg[(size_t)n * NH_ + t];
#pragma unroll
  for (int j = 0; j < 4; ++j) {
    int cidx = t + j * 128;                         // 0..511 within agg
    int comp = cidx >> 7, cc = cidx & 127;
    size_t idx = (size_t)n * NH_ + cc;
    float val;
    if (comp == 0) val = sumb[idx] / denom;
    else if (comp == 1) val = has ? mxb[idx] : 0.f;
    else if (comp == 2) val = has ? mnb[idx] : 0.f;
    else {
      float m = sumb[idx] / denom;
      float m2 = sqb[idx] / denom;
      val = sqrtf(fmaxf(m2 - m * m, 0.f) + 1e-5f);
    }
    fr[128 + cidx]  = val;
    fr[640 + cidx]  = val * r1;
    fr[1152 + cidx] = val * r2;
  }
}

__global__ void bn_stats_kernel(const float* __restrict__ g, float* __restrict__ bs,
                                float* __restrict__ bq) {
  int c = threadIdx.x;                              // 128 threads
  float s = 0.f, q = 0.f;
  for (int n = blockIdx.x; n < NN_; n += gridDim.x) {
    float v = g[(size_t)n * NH_ + c];
    s += v; q += v * v;
  }
  atomicAdd(&bs[c], s);
  atomicAdd(&bq[c], q);
}

__global__ void xg_update_kernel(const float* __restrict__ xg, const float* __restrict__ g,
                                 const float* __restrict__ bs, const float* __restrict__ bq,
                                 const float* __restrict__ bw, const float* __restrict__ bb,
                                 float* __restrict__ xg_out) {
  int idx = blockIdx.x * blockDim.x + threadIdx.x;
  if (idx >= NN_ * NH_) return;
  int c = idx & 127;
  float m = bs[c] / (float)NN_;
  float var = bq[c] / (float)NN_ - m * m;
  float v = (g[idx] - m) * rsqrtf(var + 1e-5f) * bw[c] + bb[c];
  xg_out[idx] = (xg[idx] + fmaxf(v, 0.f)) * 0.5f;
}

__global__ void fuse_gather_kernel(const float* __restrict__ cls, const float* __restrict__ xg,
                                   const int* __restrict__ srcI, const int* __restrict__ dstI,
                                   float* __restrict__ xcat) {
  int i = blockIdx.x, t = threadIdx.x;              // 128 threads
  float* r = xcat + (size_t)i * FUSED_;
  r[t]       = cls[(size_t)i * C_ + t];
  r[128 + t] = xg[(size_t)srcI[i] * NH_ + t];
  r[256 + t] = xg[(size_t)dstI[i] * NH_ + t];
}

__global__ void combine_half_kernel(const float* __restrict__ a, const float* __restrict__ b,
                                    float* __restrict__ y, size_t n) {
  size_t i = (size_t)blockIdx.x * blockDim.x + threadIdx.x;
  if (i < n) y[i] = 0.5f * (a[i] + b[i]);
}

__global__ void out_cls_kernel(const float* __restrict__ xfin, float* __restrict__ xtab_out) {
  int i = blockIdx.x, t = threadIdx.x;
  xtab_out[(size_t)i * S_ * C_ + t] = xfin[(size_t)i * FUSED_ + t];
}

__global__ void scatter_s_kernel(const float* __restrict__ xfin, const int* __restrict__ srcI,
                                 float* __restrict__ xg_out) {
  int i = blockIdx.x, t = threadIdx.x;
  xg_out[(size_t)srcI[i] * NH_ + t] = xfin[(size_t)i * FUSED_ + 128 + t];
}
__global__ void scatter_d_kernel(const float* __restrict__ xfin, const int* __restrict__ dstI,
                                 float* __restrict__ xg_out) {
  int i = blockIdx.x, t = threadIdx.x;
  xg_out[(size_t)dstI[i] * NH_ + t] = xfin[(size_t)i * FUSED_ + 256 + t];
}

// ---------------- host launch ----------------------------------------------
extern "C" void kernel_launch(void* const* d_in, const int* in_sizes, int n_in,
                              void* d_out, int out_size, void* d_ws, size_t ws_size,
                              hipStream_t stream) {
  (void)in_sizes; (void)n_in; (void)out_size; (void)ws_size;
  const float* x_tab = (const float*)d_in[0];
  const float* x_gnn = (const float*)d_in[1];
  const int*   eidx  = (const int*)d_in[2];
  const float* eattr = (const float*)d_in[3];
  const float* Wqkv  = (const float*)d_in[4];
  const float* bqkv  = (const float*)d_in[5];
  const float* Wo    = (const float*)d_in[6];
  const float* bo    = (const float*)d_in[7];
  const float* ln1w  = (const float*)d_in[8];
  const float* ln1b  = (const float*)d_in[9];
  const float* W1    = (const float*)d_in[10];
  const float* b1    = (const float*)d_in[11];
  const float* W2    = (const float*)d_in[12];
  const float* b2    = (const float*)d_in[13];
  const float* ln2w  = (const float*)d_in[14];
  const float* ln2b  = (const float*)d_in[15];
  const float* tabw  = (const float*)d_in[16];
  const float* tabb  = (const float*)d_in[17];
  const float* We    = (const float*)d_in[18];
  const float* be    = (const float*)d_in[19];
  const float* Wpre  = (const float*)d_in[20];
  const float* bpre  = (const float*)d_in[21];
  const float* Wpost = (const float*)d_in[22];
  const float* bpost = (const float*)d_in[23];
  const float* Wlin  = (const float*)d_in[24];
  const float* blin  = (const float*)d_in[25];
  const float* bnw   = (const float*)d_in[26];
  const float* bnb   = (const float*)d_in[27];
  const float* flnw  = (const float*)d_in[28];
  const float* flnb  = (const float*)d_in[29];
  const float* Wf1   = (const float*)d_in[30];
  const float* bf1   = (const float*)d_in[31];
  const float* Wf2   = (const float*)d_in[32];
  const float* bf2   = (const float*)d_in[33];
  const float* Wf3   = (const float*)d_in[34];
  const float* bf3   = (const float*)d_in[35];
  const float* fnw   = (const float*)d_in[36];
  const float* fnb   = (const float*)d_in[37];

  const int* srcI = eidx;            // edge_index[0]
  const int* dstI = eidx + EE_;      // edge_index[1]

  float* ws  = (float*)d_ws;
  float* out = (float*)d_out;
  float* out_xtab = out;
  float* out_xg   = out + (size_t)MTAB_ * C_;
  float* out_ea   = out_xg + (size_t)NN_ * NH_;

  // workspace layout (floats), with temporal reuse
  constexpr size_t SZ_QKV  = (size_t)MTAB_ * 384;
  constexpr size_t SZ_O    = (size_t)MTAB_ * C_;
  constexpr size_t SZ_NODE = (size_t)NN_ * NH_;
  constexpr size_t SZ_F    = (size_t)B_ * FUSED_;
  constexpr size_t OFF_QKV = 0;                 // later reused for e (E*128)
  constexpr size_t OFF_O   = OFF_QKV + SZ_QKV;  // later reused for feat (N*1664)
  constexpr size_t OFF_H1P = OFF_O + SZ_O;      // also h2pre; feat overflows here
  constexpr size_t OFF_H   = OFF_H1P + SZ_O;
  constexpr size_t OFF_FF  = OFF_H + SZ_O;      // also h2
  constexpr size_t OFF_CLS = OFF_FF + SZ_O;
  constexpr size_t OFF_SUM = OFF_CLS + (size_t)B_ * C_;
  constexpr size_t OFF_SQ  = OFF_SUM + SZ_NODE;
  constexpr size_t OFF_MX  = OFF_SQ + SZ_NODE;
  constexpr size_t OFF_MN  = OFF_MX + SZ_NODE;
  constexpr size_t OFF_CNT = OFF_MN + SZ_NODE;
  constexpr size_t OFF_AVG = OFF_CNT + NN_;
  constexpr size_t OFF_PST = OFF_AVG + 32;
  constexpr size_t OFF_LIN = OFF_PST + SZ_NODE;
  constexpr size_t OFF_BNS = OFF_LIN + SZ_NODE;
  constexpr size_t OFF_BNQ = OFF_BNS + 128;
  constexpr size_t OFF_XCT = OFF_BNQ + 128;
  constexpr size_t OFF_FA  = OFF_XCT + SZ_F;
  constexpr size_t OFF_FB  = OFF_FA + SZ_F;
  constexpr size_t OFF_XF  = OFF_FB + SZ_F;
  constexpr size_t OFF_E    = OFF_QKV;          // E*128 <= SZ_QKV
  constexpr size_t OFF_FEAT = OFF_O;            // N*1664 <= 2*SZ_O

  auto gemm = [&](int epi, const float* A, const float* Wm, const float* bias,
                  const float* R, float* Y, int M, int Nn, int K) {
    int blocks = ((M + 63) / 64) * (Nn / 64);
    switch (epi) {
      case 0: gemm_wmma_kernel<0><<<blocks, 128, 0, stream>>>(A, Wm, bias, R, Y, M, Nn, K); break;
      case 1: gemm_wmma_kernel<1><<<blocks, 128, 0, stream>>>(A, Wm, bias, R, Y, M, Nn, K); break;
      case 2: gemm_wmma_kernel<2><<<blocks, 128, 0, stream>>>(A, Wm, bias, R, Y, M, Nn, K); break;
      default: gemm_wmma_kernel<3><<<blocks, 128, 0, stream>>>(A, Wm, bias, R, Y, M, Nn, K); break;
    }
  };
  auto ln = [&](const float* X, const float* w, const float* b, float* Y, int rows, int width) {
    ln_kernel<<<(rows + 3) / 4, 128, 0, stream>>>(X, w, b, Y, rows, width);
  };

  // edge_attr passthrough (independent)
  copy_kernel<<<4096, 256, 0, stream>>>(eattr, out_ea, (size_t)EE_ * NH_);

  // ---- tabular transformer path ----
  gemm(EPI_BIAS, x_tab, Wqkv, bqkv, nullptr, ws + OFF_QKV, MTAB_, 384, C_);
  attn_kernel<<<B_ * H_, 32, 0, stream>>>(ws + OFF_QKV, ws + OFF_O);
  gemm(EPI_RES, ws + OFF_O, Wo, bo, x_tab, ws + OFF_H1P, MTAB_, C_, C_);
  ln(ws + OFF_H1P, ln1w, ln1b, ws + OFF_H, MTAB_, C_);
  gemm(EPI_RELU, ws + OFF_H, W1, b1, nullptr, ws + OFF_FF, MTAB_, C_, C_);
  gemm(EPI_RES, ws + OFF_FF, W2, b2, ws + OFF_H, ws + OFF_H1P, MTAB_, C_, C_); // h2pre
  ln(ws + OFF_H1P, ln2w, ln2b, ws + OFF_FF, MTAB_, C_);                        // h2
  ln(ws + OFF_FF, tabw, tabb, out_xtab, MTAB_, C_);                            // x_tab final
  cls_extract_kernel<<<B_, 128, 0, stream>>>(out_xtab, ws + OFF_CLS);

  // ---- GNN path ----
  gemm(EPI_BIAS, eattr, We, be, nullptr, ws + OFF_E, EE_, NH_, NH_);
  init_stats_kernel<<<(NN_ * NH_ + 255) / 256, 256, 0, stream>>>(
      ws + OFF_SUM, ws + OFF_SQ, ws + OFF_MX, ws + OFF_MN,
      ws + OFF_CNT, ws + OFF_AVG, ws + OFF_BNS, ws + OFF_BNQ);
  count_kernel<<<(EE_ + 255) / 256, 256, 0, stream>>>(dstI, ws + OFF_CNT);
  gemm_msg_kernel<<<((EE_ / 16) * (NH_ / 64) + 3) / 4, 128, 0, stream>>>(
      x_gnn, srcI, dstI, ws + OFF_E, Wpre, bpre,
      ws + OFF_SUM, ws + OFF_SQ, ws + OFF_MX, ws + OFF_MN);
  avg_log_kernel<<<(NN_ + 255) / 256, 256, 0, stream>>>(ws + OFF_CNT, ws + OFF_AVG);
  feat_kernel<<<NN_, 128, 0, stream>>>(x_gnn, ws + OFF_SUM, ws + OFF_SQ,
                                       ws + OFF_MX, ws + OFF_MN, ws + OFF_CNT,
                                       ws + OFF_AVG, ws + OFF_FEAT);
  gemm(EPI_BIAS, ws + OFF_FEAT, Wpost, bpost, nullptr, ws + OFF_PST, NN_, NH_, 13 * NH_);
  gemm(EPI_BIAS, ws + OFF_PST, Wlin, blin, nullptr, ws + OFF_LIN, NN_, NH_, NH_);
  bn_stats_kernel<<<512, 128, 0, stream>>>(ws + OFF_LIN, ws + OFF_BNS, ws + OFF_BNQ);
  xg_update_kernel<<<(NN_ * NH_ + 255) / 256, 256, 0, stream>>>(
      x_gnn, ws + OFF_LIN, ws + OFF_BNS, ws + OFF_BNQ, bnw, bnb, out_xg);

  // ---- fused head ----
  fuse_gather_kernel<<<B_, 128, 0, stream>>>(ws + OFF_CLS, out_xg, srcI, dstI, ws + OFF_XCT);
  ln(ws + OFF_XCT, flnw, flnb, ws + OFF_FA, B_, FUSED_);
  gemm(EPI_LEAKY, ws + OFF_FA, Wf1, bf1, nullptr, ws + OFF_FB, B_, FUSED_, FUSED_);
  gemm(EPI_LEAKY, ws + OFF_FB, Wf2, bf2, nullptr, ws + OFF_FA, B_, FUSED_, FUSED_);
  gemm(EPI_BIAS,  ws + OFF_FA, Wf3, bf3, nullptr, ws + OFF_FB, B_, FUSED_, FUSED_);
  ln(ws + OFF_FB, fnw, fnb, ws + OFF_FA, B_, FUSED_);
  combine_half_kernel<<<(int)((SZ_F + 255) / 256), 256, 0, stream>>>(
      ws + OFF_XCT, ws + OFF_FA, ws + OFF_XF, SZ_F);
  out_cls_kernel<<<B_, 128, 0, stream>>>(ws + OFF_XF, out_xtab);
  scatter_s_kernel<<<B_, 128, 0, stream>>>(ws + OFF_XF, srcI, out_xg);
  scatter_d_kernel<<<B_, 128, 0, stream>>>(ws + OFF_XF, dstI, out_xg);
}
